// MultiHeadAttention_37477884625165
// MI455X (gfx1250) — matrix operand order repaired and checked by reading, not compiled, not run on hardware
//
#include <hip/hip_runtime.h>
#include <math.h>

#define B_SZ 2
#define NQ   2048
#define NK   2048
#define DM   512
#define NH   8
#define DH   64
#define LDP  72   // padded LDS row (elements): 144B stride, conflict-free b128

typedef __bf16 bf16_t;
typedef __attribute__((ext_vector_type(8)))  __bf16 v8bf;
typedef __attribute__((ext_vector_type(16))) __bf16 v16bf;
typedef __attribute__((ext_vector_type(8)))  float  v8f;
typedef __attribute__((ext_vector_type(4)))  float  v4f;

union VFrag { v16bf v; v8bf h[2]; };

__device__ __forceinline__ v8f zero_v8f() {
  v8f z;
#pragma unroll
  for (int i = 0; i < 8; ++i) z[i] = 0.0f;
  return z;
}

__device__ __forceinline__ v8f wmma_bf16(v16bf a, v16bf b, v8f c) {
  return __builtin_amdgcn_wmma_f32_16x16x32_bf16(
      false, a, false, b, (short)0, c, false, false);
}

// gfx1250 async copy: global -> LDS, 16B per lane, tracked by ASYNCcnt.
// LDS address = low 32 bits of the generic shared pointer (ISA: addr[31:0]).
__device__ __forceinline__ void async_cp_b128(void* lds, const void* g) {
  uint32_t l = (uint32_t)(uintptr_t)lds;
  asm volatile("global_load_async_to_lds_b128 %0, %1, off"
               :: "v"(l), "v"(g) : "memory");
}
__device__ __forceinline__ void wait_async0() {
  asm volatile("s_wait_asynccnt 0x0" ::: "memory");
}

// ---------------------------------------------------------------------------
// Wt[j*512 + k] = (bf16) W[k*512 + j]
// ---------------------------------------------------------------------------
__global__ __launch_bounds__(256) void wt_convert(const float* __restrict__ W,
                                                  bf16_t* __restrict__ Wt) {
  int idx = blockIdx.x * 256 + threadIdx.x;
  int j = idx >> 9;
  int k = idx & (DM - 1);
  Wt[idx] = (bf16_t)W[(size_t)k * DM + j];
}

// ---------------------------------------------------------------------------
// Out = X(R x 512, f32) @ W + bias -> bf16. trans==1 writes V transposed/head.
// ---------------------------------------------------------------------------
__global__ __launch_bounds__(256) void qkv_gemm(const float* __restrict__ X,
                                                const bf16_t* __restrict__ Wt,
                                                const float* __restrict__ bias,
                                                bf16_t* __restrict__ Out,
                                                int trans) {
  const int wave = threadIdx.x >> 5;
  const int lane = threadIdx.x & 31;
  const int l16  = lane & 15;
  const int lhi  = lane >> 4;

  const int row0 = blockIdx.x * 128 + wave * 16;
  const int col0 = blockIdx.y * 64;

  v8f acc[4];
#pragma unroll
  for (int t = 0; t < 4; ++t) acc[t] = zero_v8f();

  const float* arow = X + (size_t)(row0 + l16) * DM + lhi * 8;

  for (int k0 = 0; k0 < DM; k0 += 32) {
    VFrag a;
    const float* ap = arow + k0;
    v4f c0 = *(const v4f*)(ap);
    v4f c1 = *(const v4f*)(ap + 4);
    v4f c2 = *(const v4f*)(ap + 16);
    v4f c3 = *(const v4f*)(ap + 20);
#pragma unroll
    for (int j = 0; j < 4; ++j) {
      a.v[j]      = (bf16_t)c0[j];
      a.v[4 + j]  = (bf16_t)c1[j];
      a.v[8 + j]  = (bf16_t)c2[j];
      a.v[12 + j] = (bf16_t)c3[j];
    }
#pragma unroll
    for (int t = 0; t < 4; ++t) {
      const bf16_t* bp =
          Wt + (size_t)(col0 + t * 16 + l16) * DM + k0 + lhi * 8;
      VFrag b;
      b.h[0] = *(const v8bf*)(bp);
      b.h[1] = *(const v8bf*)(bp + 16);
      acc[t] = wmma_bf16(a.v, b.v, acc[t]);
    }
  }

#pragma unroll
  for (int t = 0; t < 4; ++t) {
    const int col = col0 + t * 16 + l16;
    const float bv = bias[col];
#pragma unroll
    for (int r = 0; r < 8; ++r) {
      const int row = row0 + r + lhi * 8;
      const bf16_t ob = (bf16_t)(acc[t][r] + bv);
      if (!trans) {
        Out[(size_t)row * DM + col] = ob;
      } else {
        const int b = row >> 11;
        const int m = row & (NK - 1);
        const int h = col >> 6;
        const int c = col & (DH - 1);
        Out[((((size_t)(b * NH + h)) * DH + c) << 11) + m] = ob;
      }
    }
  }
}

// ---------------------------------------------------------------------------
// Flash attention, async double-buffered K/V staging in LDS.
// Block = 128 thr (4 waves); wave owns 16 Q rows. Grid (B*H, N/64).
// ---------------------------------------------------------------------------
__global__ __launch_bounds__(128) void attn_kernel(
    const bf16_t* __restrict__ Qbf,  // (B*N, 512)
    const bf16_t* __restrict__ Kbf,  // (B*M, 512)
    const bf16_t* __restrict__ Vt,   // (B,H,64,M)
    const float* __restrict__ af,    // (B,N,M)
    const float* __restrict__ kw,    // (B,M)
    float* __restrict__ Out) {       // (B,N,512)
  __shared__ __align__(16) bf16_t Kbuf[2][64 * LDP];  // 9 KB x2
  __shared__ __align__(16) bf16_t Vbuf[2][64 * LDP];  // 9 KB x2
  __shared__ __align__(16) bf16_t Pbuf[4][16 * LDP];  // 9 KB

  const int tid  = threadIdx.x;
  const int wave = tid >> 5;
  const int lane = tid & 31;
  const int l16  = lane & 15;
  const int lhi  = lane >> 4;

  const int bh = blockIdx.x;
  const int b  = bh >> 3;
  const int h  = bh & (NH - 1);
  const int n0 = blockIdx.y * 64 + wave * 16;

  // Per-(b,h) global bases for the staged chunks
  const bf16_t* Kg = Kbf + (size_t)b * NK * DM + h * DH;    // + (m0+row)*DM + off
  const bf16_t* Vg = Vt + (((size_t)bh * DH) << 11);        // + (row<<11) + m0 + off

  // ---- Q A-fragments (loaded once) ----------------------------------------
  const bf16_t* qrow = Qbf + ((size_t)b * NQ + n0 + l16) * DM + h * DH;
  VFrag qa[2];
#pragma unroll
  for (int kk = 0; kk < 2; ++kk) {
    qa[kk].h[0] = *(const v8bf*)(qrow + kk * 32 + lhi * 8);
    qa[kk].h[1] = *(const v8bf*)(qrow + kk * 32 + 16 + lhi * 8);
  }

  v8f o[4];
#pragma unroll
  for (int t = 0; t < 4; ++t) o[t] = zero_v8f();

  float mrow[8], lrow[8];
#pragma unroll
  for (int r = 0; r < 8; ++r) { mrow[r] = -3.0e38f; lrow[r] = 0.0f; }

  const float* afb = af + (size_t)b * NQ * NK;
  const float* kwb = kw + (size_t)b * NK;
  bf16_t* pb = &Pbuf[wave][0];

  // ---- prologue: stage chunk 0 into buffer 0 ------------------------------
#pragma unroll
  for (int i = 0; i < 4; ++i) {
    const int s   = tid + i * 128;       // 512 x 16B segments each for K and V
    const int row = s >> 3;
    const int off = (s & 7) * 8;
    async_cp_b128(&Kbuf[0][row * LDP + off], Kg + (size_t)row * DM + off);
    async_cp_b128(&Vbuf[0][row * LDP + off], Vg + ((size_t)row << 11) + off);
  }

  int buf = 0;
  for (int m0 = 0; m0 < NK; m0 += 64) {
    wait_async0();        // this wave's copies into `buf` are done
    __syncthreads();      // all waves' copies done; prior reads of buf^1 done

    if (m0 + 64 < NK) {   // stage next chunk into the other buffer
#pragma unroll
      for (int i = 0; i < 4; ++i) {
        const int s   = tid + i * 128;
        const int row = s >> 3;
        const int off = (s & 7) * 8;
        async_cp_b128(&Kbuf[buf ^ 1][row * LDP + off],
                      Kg + (size_t)(m0 + 64 + row) * DM + off);
        async_cp_b128(&Vbuf[buf ^ 1][row * LDP + off],
                      Vg + ((size_t)row << 11) + m0 + 64 + off);
      }
    }

    // ---- S = Q K^T from LDS ----------------------------------------------
    const bf16_t* kb = &Kbuf[buf][0];
    v8f s4[4];
#pragma unroll
    for (int t = 0; t < 4; ++t) {
      const bf16_t* kp = kb + (t * 16 + l16) * LDP;
      v8f acc = zero_v8f();
#pragma unroll
      for (int kk = 0; kk < 2; ++kk) {
        VFrag kf;
        kf.h[0] = *(const v8bf*)(kp + kk * 32 + lhi * 8);
        kf.h[1] = *(const v8bf*)(kp + kk * 32 + 16 + lhi * 8);
        acc = wmma_bf16(qa[kk].v, kf.v, acc);
      }
      s4[t] = acc;
    }

    // prefetch next chunk of the dominant attention_factors stream
    if (m0 + 64 < NK) {
#pragma unroll
      for (int r = 0; r < 8; ++r) {
        const int n = n0 + r + lhi * 8;
        __builtin_prefetch(&afb[(size_t)n * NK + m0 + 64 + l16 * 4], 0, 0);
      }
    }

    // ---- scale by 1/8 * attention_factors * key_weights -------------------
    float cmax[8];
#pragma unroll
    for (int r = 0; r < 8; ++r) cmax[r] = -3.0e38f;
#pragma unroll
    for (int t = 0; t < 4; ++t) {
      const int m = m0 + t * 16 + l16;
      const float kwv = kwb[m];
#pragma unroll
      for (int r = 0; r < 8; ++r) {
        const int n = n0 + r + lhi * 8;
        const float a_ = afb[(size_t)n * NK + m];
        const float val = s4[t][r] * 0.125f * a_ * kwv;
        s4[t][r] = val;
        cmax[r] = fmaxf(cmax[r], val);
      }
    }
#pragma unroll
    for (int off = 1; off < 16; off <<= 1) {
#pragma unroll
      for (int r = 0; r < 8; ++r)
        cmax[r] = fmaxf(cmax[r], __shfl_xor(cmax[r], off, 32));
    }

    // ---- online softmax ---------------------------------------------------
    float alpha[8];
#pragma unroll
    for (int r = 0; r < 8; ++r) {
      const float mn = fmaxf(mrow[r], cmax[r]);
      alpha[r] = __expf(mrow[r] - mn);
      mrow[r] = mn;
      lrow[r] *= alpha[r];
    }
#pragma unroll
    for (int t = 0; t < 4; ++t)
#pragma unroll
      for (int r = 0; r < 8; ++r) o[t][r] *= alpha[r];

    float prow[8];
#pragma unroll
    for (int r = 0; r < 8; ++r) prow[r] = 0.0f;
#pragma unroll
    for (int t = 0; t < 4; ++t) {
#pragma unroll
      for (int r = 0; r < 8; ++r) {
        const float p = __expf(s4[t][r] - mrow[r]);
        prow[r] += p;
        pb[(r + lhi * 8) * LDP + t * 16 + l16] = (bf16_t)p;
      }
    }
#pragma unroll
    for (int off = 1; off < 16; off <<= 1) {
#pragma unroll
      for (int r = 0; r < 8; ++r) prow[r] += __shfl_xor(prow[r], off, 32);
    }
#pragma unroll
    for (int r = 0; r < 8; ++r) lrow[r] += prow[r];

    __syncthreads();  // P visible; (uniform across waves)

    // ---- O += P @ V from LDS ---------------------------------------------
    const bf16_t* vb = &Vbuf[buf][0];
#pragma unroll
    for (int kk = 0; kk < 2; ++kk) {
      VFrag pa;
      const bf16_t* pr = pb + l16 * LDP + kk * 32;
      pa.h[0] = *(const v8bf*)(pr + lhi * 8);
      pa.h[1] = *(const v8bf*)(pr + 16 + lhi * 8);
#pragma unroll
      for (int t = 0; t < 4; ++t) {
        const bf16_t* vp = vb + (t * 16 + l16) * LDP + kk * 32 + lhi * 8;
        VFrag vf;
        vf.h[0] = *(const v8bf*)(vp);
        vf.h[1] = *(const v8bf*)(vp + 16);
        o[t] = wmma_bf16(pa.v, vf.v, o[t]);
      }
    }
    buf ^= 1;
  }

  // ---- normalize and write output -----------------------------------------
  float inv[8];
#pragma unroll
  for (int r = 0; r < 8; ++r) inv[r] = 1.0f / lrow[r];
#pragma unroll
  for (int t = 0; t < 4; ++t) {
#pragma unroll
    for (int r = 0; r < 8; ++r) {
      const int n = n0 + r + lhi * 8;
      const int c = h * DH + t * 16 + l16;
      Out[((size_t)b * NQ + n) * DM + c] = o[t][r] * inv[r];
    }
  }
}

// ---------------------------------------------------------------------------
extern "C" void kernel_launch(void* const* d_in, const int* in_sizes, int n_in,
                              void* d_out, int out_size, void* d_ws,
                              size_t ws_size, hipStream_t stream) {
  const float* xq = (const float*)d_in[0];
  const float* xk = (const float*)d_in[1];
  const float* xv = (const float*)d_in[2];
  const float* kw = (const float*)d_in[3];
  const float* af = (const float*)d_in[4];
  const float* Wq = (const float*)d_in[5];
  const float* bq = (const float*)d_in[6];
  const float* Wk = (const float*)d_in[7];
  const float* bk = (const float*)d_in[8];
  const float* Wv = (const float*)d_in[9];
  const float* bv = (const float*)d_in[10];
  float* out = (float*)d_out;

  char* ws = (char*)d_ws;
  bf16_t* Qbf = (bf16_t*)(ws);                 // 4 MiB
  bf16_t* Kbf = (bf16_t*)(ws + (4u << 20));    // 4 MiB
  bf16_t* Vt  = (bf16_t*)(ws + (8u << 20));    // 4 MiB (B,H,64,M)
  bf16_t* Wqt = (bf16_t*)(ws + (12u << 20));   // 3 x 0.5 MiB
  bf16_t* Wkt = Wqt + DM * DM;
  bf16_t* Wvt = Wkt + DM * DM;

  dim3 wtg(DM * DM / 256);
  wt_convert<<<wtg, 256, 0, stream>>>(Wq, Wqt);
  wt_convert<<<wtg, 256, 0, stream>>>(Wk, Wkt);
  wt_convert<<<wtg, 256, 0, stream>>>(Wv, Wvt);

  dim3 gg(32, 8);
  qkv_gemm<<<gg, 256, 0, stream>>>(xq, Wqt, bq, Qbf, 0);
  qkv_gemm<<<gg, 256, 0, stream>>>(xk, Wkt, bk, Kbf, 0);
  qkv_gemm<<<gg, 256, 0, stream>>>(xv, Wvt, bv, Vt, 1);

  dim3 ag(B_SZ * NH, NQ / 64);
  attn_kernel<<<ag, 128, 0, stream>>>(Qbf, Kbf, Vt, af, kw, out);
}
